// StandardLSTM_62637803045438
// MI455X (gfx1250) — compile-verified, matrix operand order checked by reading, not run on hardware
//
#include <hip/hip_runtime.h>

typedef float   v2f  __attribute__((ext_vector_type(2)));
typedef float   v8f  __attribute__((ext_vector_type(8)));
typedef _Float16 v16h __attribute__((ext_vector_type(16)));

#define B_ 2048
#define T_ 8192
#define CHUNK 32
#define NCHUNK (T_ / CHUNK)
#define NW 108   // total weight count

// ---------------- fast activation helpers (CDNA5 v_tanh_f32 if available) ----------
__device__ __forceinline__ float fast_tanh(float x) {
#if __has_builtin(__builtin_amdgcn_tanhf)
  return __builtin_amdgcn_tanhf(x);
#else
  // tanh(x) = 2/(1+e^{-2x}) - 1 ; e^{-2x} = exp2(-2*log2(e)*x)
  float e = __builtin_amdgcn_exp2f(x * -2.8853900817779268f);
  return __builtin_amdgcn_rcpf(1.0f + e) * 2.0f - 1.0f;
#endif
}
// input already pre-scaled by 0.5: sigmoid(a) = 0.5*tanh(a/2)+0.5
__device__ __forceinline__ float sig_half(float xh) {
  return __builtin_fmaf(fast_tanh(xh), 0.5f, 0.5f);
}

// scaled recurrent weight: cols [i0..i2,f0..f2,g0..g2,o0..o2]; sigmoid gates *0.5
__device__ __forceinline__ float wh_scaled(const float* w, int k, int col) {
  int g = col / 3, j = col % 3;
  float v = w[36 + g * 9 + k * 3 + j];
  return (g == 2) ? v : v * 0.5f;
}

// ---------------- main LSTM kernel: 1 lane = 1 batch row, 1 wave per block --------
__global__ __launch_bounds__(32) void lstm_kernel(const int* __restrict__ x,
                                                  const float* __restrict__ w,
                                                  float* __restrict__ out) {
  __shared__ alignas(16) float tab[36];            // 3 symbols x 12 pre-scaled bases
  __shared__ alignas(16) int   xs[2 * CHUNK * 32]; // double-buffered x staging (8 KB)

  const int lane = (int)threadIdx.x;               // 0..31
  const int b    = (int)blockIdx.x * 32 + lane;    // batch row

  // Build base table: tab[s*12+col] = scale*(W_i[g][s][j] + b_i[g][j] + b_h[g][j])
  for (int i = lane; i < 36; i += 32) {
    int s = i / 12, col = i % 12, g = col / 3, j = col % 3;
    float v = w[g * 9 + s * 3 + j] + w[72 + g * 3 + j] + w[84 + g * 3 + j];
    if (g != 2) v *= 0.5f;
    tab[s * 12 + col] = v;
  }
  __syncthreads();

  // Recurrent weights as packed pairs (wave-uniform): Wp[k][p] covers cols (2p,2p+1)
  v2f Wp[3][6];
#pragma unroll
  for (int k = 0; k < 3; ++k)
#pragma unroll
    for (int p = 0; p < 6; ++p)
      Wp[k][p] = (v2f){wh_scaled(w, k, 2 * p), wh_scaled(w, k, 2 * p + 1)};

  float Wo[3][3], bo[3];
#pragma unroll
  for (int k = 0; k < 3; ++k)
#pragma unroll
    for (int o = 0; o < 3; ++o) Wo[k][o] = w[96 + k * 3 + o];
#pragma unroll
  for (int o = 0; o < 3; ++o) bo[o] = w[105 + o];

  const int*  xrow = x + (size_t)b * T_;
  float*      orow = out + (size_t)b * T_ * 3;
  const float4* tab4 = reinterpret_cast<const float4*>(tab);

  // Issue one chunk of x (32 steps) as async copies into LDS buffer (c&1).
  auto issue_chunk = [&](int c) {
    unsigned ldsBase = (unsigned)(size_t)(xs + (c & 1) * (CHUNK * 32));
    const int* gp = xrow + c * CHUNK;
#pragma unroll
    for (int g = 0; g < CHUNK / 4; ++g) {
      unsigned  la = ldsBase + (unsigned)(g * 512 + lane * 16);
      const int* ga = gp + g * 4;
      asm volatile("global_load_async_to_lds_b128 %0, %1, off"
                   :: "v"(la), "v"(ga) : "memory");
    }
  };

  issue_chunk(0);
  issue_chunk(1);

  float h0 = 0.f, h1 = 0.f, h2 = 0.f;
  float cs0 = 0.f, cs1 = 0.f, cs2 = 0.f;

  for (int c = 0; c < NCHUNK; ++c) {
    if (c + 1 < NCHUNK)
      asm volatile("s_wait_asynccnt 8" ::: "memory");   // oldest chunk complete
    else
      asm volatile("s_wait_asynccnt 0" ::: "memory");   // last chunk: drain all

    const int* xls = xs + (c & 1) * (CHUNK * 32);

    for (int g = 0; g < CHUNK / 4; ++g) {
      int4 xv = *reinterpret_cast<const int4*>(&xls[g * 128 + lane * 4]);
      int xa[4] = {xv.x, xv.y, xv.z, xv.w};

      // prefetch all 4 steps' base rows up-front (only depend on x, not h)
      float4 q[4][3];
#pragma unroll
      for (int u = 0; u < 4; ++u) {
        const float4* tp = tab4 + xa[u] * 3;
        q[u][0] = tp[0]; q[u][1] = tp[1]; q[u][2] = tp[2];
      }

      float res[12];
#pragma unroll
      for (int u = 0; u < 4; ++u) {
        v2f a0 = {q[u][0].x, q[u][0].y};   // (i0,i1)
        v2f a1 = {q[u][0].z, q[u][0].w};   // (i2,f0)
        v2f a2 = {q[u][1].x, q[u][1].y};   // (f1,f2)
        v2f a3 = {q[u][1].z, q[u][1].w};   // (g0,g1)
        v2f a4 = {q[u][2].x, q[u][2].y};   // (g2,o0)
        v2f a5 = {q[u][2].z, q[u][2].w};   // (o1,o2)
        v2f hh0 = {h0, h0}, hh1 = {h1, h1}, hh2 = {h2, h2};
        a0 += hh0 * Wp[0][0] + hh1 * Wp[1][0] + hh2 * Wp[2][0];
        a1 += hh0 * Wp[0][1] + hh1 * Wp[1][1] + hh2 * Wp[2][1];
        a2 += hh0 * Wp[0][2] + hh1 * Wp[1][2] + hh2 * Wp[2][2];
        a3 += hh0 * Wp[0][3] + hh1 * Wp[1][3] + hh2 * Wp[2][3];
        a4 += hh0 * Wp[0][4] + hh1 * Wp[1][4] + hh2 * Wp[2][4];
        a5 += hh0 * Wp[0][5] + hh1 * Wp[1][5] + hh2 * Wp[2][5];

        float it0 = sig_half(a0.x), it1 = sig_half(a0.y), it2 = sig_half(a1.x);
        float ft0 = sig_half(a1.y), ft1 = sig_half(a2.x), ft2 = sig_half(a2.y);
        float gg0 = fast_tanh(a3.x), gg1 = fast_tanh(a3.y), gg2 = fast_tanh(a4.x);
        float ot0 = sig_half(a4.y), ot1 = sig_half(a5.x), ot2 = sig_half(a5.y);

        cs0 = ft0 * cs0 + it0 * gg0;
        cs1 = ft1 * cs1 + it1 * gg1;
        cs2 = ft2 * cs2 + it2 * gg2;
        h0 = ot0 * fast_tanh(cs0);
        h1 = ot1 * fast_tanh(cs1);
        h2 = ot2 * fast_tanh(cs2);

        res[3 * u + 0] = bo[0] + h0 * Wo[0][0] + h1 * Wo[1][0] + h2 * Wo[2][0];
        res[3 * u + 1] = bo[1] + h0 * Wo[0][1] + h1 * Wo[1][1] + h2 * Wo[2][1];
        res[3 * u + 2] = bo[2] + h0 * Wo[0][2] + h1 * Wo[1][2] + h2 * Wo[2][2];
      }

      float4* op = reinterpret_cast<float4*>(orow + (size_t)(c * CHUNK + g * 4) * 3);
      op[0] = make_float4(res[0], res[1], res[2],  res[3]);
      op[1] = make_float4(res[4], res[5], res[6],  res[7]);
      op[2] = make_float4(res[8], res[9], res[10], res[11]);
    }

    if (c + 2 < NCHUNK) issue_chunk(c + 2);  // refill the buffer we just drained
  }
}

// ---------------- l1/l2 regularization via WMMA row-sum reduction ------------------
__global__ __launch_bounds__(32) void reg_kernel(const float* __restrict__ w,
                                                 float* __restrict__ out) {
  const int lane = (int)threadIdx.x;
  const int m = lane & 15;
  float l1, l2;
#if __has_builtin(__builtin_amdgcn_wmma_f32_16x16x4_f32)
  // A(16x4) f32 layout: lanes 0-15 hold K=0,1 in v0,v1; lanes 16-31 hold K=2,3.
  const int kh = (lane >> 4) * 2;
  v2f ones2 = {1.0f, 1.0f};
  v2f aab[2], asq[2];
#pragma unroll
  for (int t = 0; t < 2; ++t) {
    int e0 = t * 64 + m * 4 + kh;
    float w0 = (e0     < NW) ? w[e0]     : 0.0f;
    float w1 = (e0 + 1 < NW) ? w[e0 + 1] : 0.0f;
    aab[t] = (v2f){fabsf(w0), fabsf(w1)};
    asq[t] = (v2f){w0 * w0, w1 * w1};
  }
  v8f d1 = {};
  d1 = __builtin_amdgcn_wmma_f32_16x16x4_f32(false, aab[0], false, ones2, 0, d1, false, false);
  d1 = __builtin_amdgcn_wmma_f32_16x16x4_f32(false, aab[1], false, ones2, 0, d1, false, false);
  v8f d2 = {};
  d2 = __builtin_amdgcn_wmma_f32_16x16x4_f32(false, asq[0], false, ones2, 0, d2, false, false);
  d2 = __builtin_amdgcn_wmma_f32_16x16x4_f32(false, asq[1], false, ones2, 0, d2, false, false);
#else
  // Fallback: probe-confirmed f16 WMMA. A(16x32): one tile holds all 512 slots.
  const int hi = lane >> 4;
  v16h aab, asq, onesh;
#pragma unroll
  for (int i = 0; i < 16; ++i) {
    int K = ((i < 8) ? i : (i + 8)) + hi * 8;
    int e = m * 32 + K;
    float v = (e < NW) ? w[e] : 0.0f;
    aab[i] = (_Float16)fabsf(v);
    asq[i] = (_Float16)(v * v);
    onesh[i] = (_Float16)1.0f;
  }
  v8f d1 = {}, d2 = {};
  d1 = __builtin_amdgcn_wmma_f32_16x16x32_f16(false, aab, false, onesh, (short)0, d1, false, false);
  d2 = __builtin_amdgcn_wmma_f32_16x16x32_f16(false, asq, false, onesh, (short)0, d2, false, false);
#endif
  // D[m][n] = rowsum(m) for every n. Sum the 8 VGPRs per lane, then combine
  // lane 0 (rows 0-7) with lane 16 (rows 8-15).
  float s1 = 0.f, s2 = 0.f;
#pragma unroll
  for (int i = 0; i < 8; ++i) { s1 += d1[i]; s2 += d2[i]; }
  l1 = __shfl(s1, 0, 32) + __shfl(s1, 16, 32);
  l2 = __shfl(s2, 0, 32) + __shfl(s2, 16, 32);
  if (lane == 0) {
    out[(size_t)B_ * T_ * 3 + 0] = l1;
    out[(size_t)B_ * T_ * 3 + 1] = l2;
  }
}

extern "C" void kernel_launch(void* const* d_in, const int* in_sizes, int n_in,
                              void* d_out, int out_size, void* d_ws, size_t ws_size,
                              hipStream_t stream) {
  (void)in_sizes; (void)n_in; (void)d_ws; (void)ws_size; (void)out_size;
  const int*   x = (const int*)d_in[0];
  const float* w = (const float*)d_in[1];
  float*     out = (float*)d_out;

  lstm_kernel<<<dim3(B_ / 32), dim3(32), 0, stream>>>(x, w, out);
  reg_kernel<<<dim3(1), dim3(32), 0, stream>>>(w, out);
}
